// BailingMoELinearAttention_89429809037999
// MI455X (gfx1250) — matrix operand order repaired and checked by reading, not compiled, not run on hardware
//
#include <hip/hip_runtime.h>

// ---------------------------------------------------------------------------
// Feature gates
// ---------------------------------------------------------------------------
#if defined(__has_builtin)
#if __has_builtin(__builtin_amdgcn_global_load_async_to_lds_b128) && \
    __has_builtin(__builtin_amdgcn_s_wait_asynccnt)
#define HAVE_ASYNC_LDS 1
#endif
#if __has_builtin(__builtin_amdgcn_s_wait_tensorcnt)
#define HAVE_TENSORCNT_BUILTIN 1
#endif
#endif

#ifndef TRY_TDM
#define TRY_TDM 1   // inline-asm tensor_load_to_lds path in phaseC
#endif

// ---------------------------------------------------------------------------
// Types / helpers
// ---------------------------------------------------------------------------
typedef unsigned short u16;
typedef unsigned int   u32;
typedef __attribute__((ext_vector_type(16))) __bf16 bf16x16;
typedef __attribute__((ext_vector_type(8)))  __bf16 bf16x8;
typedef __attribute__((ext_vector_type(8)))  float  f32x8;
typedef __attribute__((ext_vector_type(4)))  u32    u32x4;
typedef __attribute__((ext_vector_type(8)))  u32    u32x8;
typedef int i32x4v __attribute__((vector_size(16)));   // matches async-LDS builtin params

__device__ __forceinline__ float bf2f(u16 x) {
    u32 u = ((u32)x) << 16;
    return __builtin_bit_cast(float, u);
}
__device__ __forceinline__ u16 f2bf(float f) {   // round-to-nearest-even
    u32 u = __builtin_bit_cast(u32, f);
    u32 r = u + 0x7FFFu + ((u >> 16) & 1u);
    return (u16)(r >> 16);
}
__device__ __forceinline__ bf16x16 cat8(bf16x8 lo, bf16x8 hi) {
    return __builtin_shufflevector(lo, hi, 0,1,2,3,4,5,6,7,8,9,10,11,12,13,14,15);
}
// A-operand fragment: lane holds 16 contiguous elems at p[0..15]
__device__ __forceinline__ bf16x16 load_fragA(const u16* p) {
    bf16x8 lo = *(const bf16x8*)p;
    bf16x8 hi = *(const bf16x8*)(p + 8);
    return cat8(lo, hi);
}
// B-operand fragment: lane holds 16 contiguous elems at p[0..15]
__device__ __forceinline__ bf16x16 load_fragB(const u16* p) {
    bf16x8 lo = *(const bf16x8*)p;
    bf16x8 hi = *(const bf16x8*)(p + 8);
    return cat8(lo, hi);
}
__device__ __forceinline__ f32x8 wmma_bf16(bf16x16 a, bf16x16 b, f32x8 c) {
    return __builtin_amdgcn_wmma_f32_16x16x32_bf16(false, a, false, b,
                                                   (short)0, c, false, false);
}
// ALiBi-style slopes for H=16, layer 0 of 32: 2^{-0.5(h+1)} * (1+1e-5)
__device__ __forceinline__ float slope_h(int h) {
    return __expf(-0.34657359027997264f * (float)(h + 1)) * 1.00001f;
}

// --- async global->LDS copy (16 bytes / lane), with portable fallback -------
__device__ __forceinline__ void async_copy16(const u16* g, u16* s) {
#if HAVE_ASYNC_LDS
    __builtin_amdgcn_global_load_async_to_lds_b128(
        (__attribute__((address_space(1))) i32x4v*)g,
        (__attribute__((address_space(3))) i32x4v*)s, 0, 0);
#else
    *(uint4*)s = *(const uint4*)g;
#endif
}
__device__ __forceinline__ void async_join() {
#if HAVE_ASYNC_LDS
    __builtin_amdgcn_s_wait_asynccnt(0);
#endif
}

// --- TDM: 2D tile load, global -> LDS, bf16 elements ------------------------
__device__ __forceinline__ u32 lds_off_u32(const void* p) {
    return (u32)(size_t)(__attribute__((address_space(3))) const void*)p;
}
__device__ __forceinline__ void tensor_wait0() {
#if HAVE_TENSORCNT_BUILTIN
    __builtin_amdgcn_s_wait_tensorcnt(0);
#else
    asm volatile("s_wait_tensorcnt 0x0" ::: "memory");
#endif
}
#if TRY_TDM
// tile_w elems wide (dim0), tile_h rows (dim1), row stride in elems,
// LDS row padding expressed via D# pad fields (interval/amount codes).
__device__ __forceinline__ void tdm_load_2d_bf16(u32 lds_off, const void* gptr,
                                                 u32 tile_w, u32 tile_h,
                                                 u32 stride_elems,
                                                 u32 pad_int_code, u32 pad_amt_code) {
    unsigned long long ga = (unsigned long long)gptr;
    u32x4 g0;
    g0[0] = 1u;                               // count=1, user descriptor
    g0[1] = lds_off;                          // lds_addr (bytes)
    g0[2] = (u32)ga;                          // global_addr[31:0]
    g0[3] = (u32)(ga >> 32) | (2u << 30);     // global_addr[56:32] | type=2
    u32 tdim0 = stride_elems;                 // no OOB along dim0
    u32 tdim1 = 1u << 20;                     // no OOB along dim1
    u32x8 g1;
    g1[0] = (1u << 16)                        // data_size = 2 bytes
          | (1u << 20)                        // pad_enable
          | (pad_int_code << 22)
          | (pad_amt_code << 25);
    g1[1] = (tdim0 & 0xFFFFu) << 16;                       // tensor_dim0 lo
    g1[2] = (tdim0 >> 16) | ((tdim1 & 0xFFFFu) << 16);     // dim0 hi | dim1 lo
    g1[3] = (tdim1 >> 16) | (tile_w << 16);                // dim1 hi | tile_dim0
    g1[4] = tile_h;                                        // tile_dim1 (tile_dim2=0)
    g1[5] = stride_elems;                                  // tensor_dim0_stride lo
    g1[6] = 0u;                                            // stride hi | dim1_stride lo
    g1[7] = 0u;
    asm volatile("tensor_load_to_lds %0, %1" :: "s"(g0), "s"(g1) : "memory");
}
#endif

// ---------------------------------------------------------------------------
// Problem constants
// ---------------------------------------------------------------------------
constexpr int T   = 4096;
constexpr int HID = 2048;
constexpr int H   = 16;
constexpr int D   = 128;
constexpr int BLK = 256;
constexpr int NC  = T / BLK;   // 16

// ---------------------------------------------------------------------------
// f32 -> bf16 elementwise convert
// ---------------------------------------------------------------------------
__global__ __launch_bounds__(256) void cvt_bf16(const float* __restrict__ in,
                                                u16* __restrict__ out, int n) {
    int i = (blockIdx.x * 256 + threadIdx.x) * 4;
    if (i < n) {
        float4 v = *(const float4*)(in + i);
        out[i + 0] = f2bf(v.x);
        out[i + 1] = f2bf(v.y);
        out[i + 2] = f2bf(v.z);
        out[i + 3] = f2bf(v.w);
    }
}

// ---------------------------------------------------------------------------
// f32 [R][C] -> bf16 [C][R] tiled transpose (weights)
// ---------------------------------------------------------------------------
__global__ __launch_bounds__(256) void transpose_cvt(const float* __restrict__ in,
                                                     u16* __restrict__ out,
                                                     int R, int C) {
    __shared__ u16 tile[32][33];
    int bx = blockIdx.x * 32;
    int by = blockIdx.y * 32;
    int tx = threadIdx.x & 31;
    int ty = threadIdx.x >> 5;
#pragma unroll
    for (int p = 0; p < 4; ++p) {
        int r = by + ty + p * 8;
        tile[ty + p * 8][tx] = f2bf(in[(size_t)r * C + bx + tx]);
    }
    __syncthreads();
#pragma unroll
    for (int p = 0; p < 4; ++p) {
        int cc = bx + ty + p * 8;
        out[(size_t)cc * R + by + tx] = tile[tx][ty + p * 8];
    }
}

// ---------------------------------------------------------------------------
// NT WMMA GEMM: C[M][N] = A[M][K] * Bt[N][K]^T (bf16 in, f32 acc).
// 256 thr / 8 waves, tile 128x128, BK=64, double-buffered LDS with
// async global->LDS fills overlapping the WMMAs.
// ---------------------------------------------------------------------------
template <bool OUT_BF16>
__global__ __launch_bounds__(256) void gemm_nt(const u16* __restrict__ A,
                                               const u16* __restrict__ Bt,
                                               void* __restrict__ Cout,
                                               int M, int N, int K) {
    __shared__ u16 ldsA[2][128 * 72];
    __shared__ u16 ldsB[2][128 * 72];
    const int bm = blockIdx.y * 128, bn = blockIdx.x * 128;
    const int tid = threadIdx.x;
    const int wave = tid >> 5, lane = tid & 31, l16 = lane & 15;
    const int khalf = (lane >> 4) & 1;
    const int wm = (wave >> 2) * 64;
    const int wn = (wave & 3) * 32;

    f32x8 acc[4][2] = {};

    const int lrow = tid >> 1;          // 0..127
    const int lcol = (tid & 1) * 32;    // 0 / 32
    const u16* gA = A + (size_t)(bm + lrow) * K + lcol;
    const u16* gB = Bt + (size_t)(bn + lrow) * K + lcol;

    auto fill = [&](int buf, int k0) {
        const u16* ga = gA + k0;
        const u16* gb = gB + k0;
        u16* sa = &ldsA[buf][lrow * 72 + lcol];
        u16* sb = &ldsB[buf][lrow * 72 + lcol];
#pragma unroll
        for (int e = 0; e < 4; ++e) {
            async_copy16(ga + e * 8, sa + e * 8);
            async_copy16(gb + e * 8, sb + e * 8);
        }
    };

    const int nstage = K / 64;
    fill(0, 0);
    async_join();
    __syncthreads();
    for (int st = 0; st < nstage; ++st) {
        int buf = st & 1;
        if (st + 1 < nstage) fill(buf ^ 1, (st + 1) * 64);
#pragma unroll
        for (int kc = 0; kc < 2; ++kc) {
            int ko = kc * 32;
            bf16x16 bfr[2];
#pragma unroll
            for (int j = 0; j < 2; ++j)
                bfr[j] = load_fragB(&ldsB[buf][(wn + j * 16 + l16) * 72 + ko + (khalf ? 16 : 0)]);
#pragma unroll
            for (int i = 0; i < 4; ++i) {
                bf16x16 afr = load_fragA(&ldsA[buf][(wm + i * 16 + l16) * 72 + ko + (khalf ? 16 : 0)]);
#pragma unroll
                for (int j = 0; j < 2; ++j)
                    acc[i][j] = wmma_bf16(afr, bfr[j], acc[i][j]);
            }
        }
        async_join();
        __syncthreads();
    }
#pragma unroll
    for (int i = 0; i < 4; ++i)
#pragma unroll
        for (int j = 0; j < 2; ++j) {
            int col = bn + wn + j * 16 + l16;
#pragma unroll
            for (int r = 0; r < 8; ++r) {
                int row = bm + wm + i * 16 + r + khalf * 8;
                float v = acc[i][j][r];
                if (OUT_BF16)
                    ((u16*)Cout)[(size_t)row * N + col] = f2bf(v);
                else
                    ((float*)Cout)[(size_t)row * N + col] = v;
            }
        }
}

// ---------------------------------------------------------------------------
// RoPE + repack. qkv bf16 [T][3*H*D] -> q [T][H*D]*D^-0.5, k [T][H*D],
// kT/vT [NC][H][D][BLK].  One thread per (t, h, d<64).
// ---------------------------------------------------------------------------
__global__ __launch_bounds__(256) void rope_pack(const u16* __restrict__ qkv,
                                                 const int* __restrict__ positions,
                                                 u16* __restrict__ qout,
                                                 u16* __restrict__ kout,
                                                 u16* __restrict__ kTout,
                                                 u16* __restrict__ vTout) {
    int n = blockIdx.x * 256 + threadIdx.x;       // T*H*64 threads
    int t = n >> 10;
    int rem = n & 1023;
    int h = rem >> 6;
    int d = rem & 63;
    float pos = (float)positions[t];
    float inv = __expf(-(float)d * (13.304684934198283f / 64.0f)); // THETA^{-2d/D}
    float fr = pos * inv;
    float sn, cs;
    __sincosf(fr, &sn, &cs);

    const u16* row = qkv + (size_t)t * (3 * HID);
    float q1 = bf2f(row[h * 128 + d]);
    float q2 = bf2f(row[h * 128 + d + 64]);
    float k1 = bf2f(row[HID + h * 128 + d]);
    float k2 = bf2f(row[HID + h * 128 + d + 64]);
    u16 v1 = row[2 * HID + h * 128 + d];
    u16 v2 = row[2 * HID + h * 128 + d + 64];

    const float qs = 0.08838834764831845f;        // D^-0.5
    float qo1 = (q1 * cs - q2 * sn) * qs;
    float qo2 = (q1 * sn + q2 * cs) * qs;
    float ko1 = k1 * cs - k2 * sn;
    float ko2 = k1 * sn + k2 * cs;

    size_t ob = (size_t)t * HID + h * 128 + d;
    qout[ob] = f2bf(qo1); qout[ob + 64] = f2bf(qo2);
    kout[ob] = f2bf(ko1); kout[ob + 64] = f2bf(ko2);

    int c = t >> 8, j = t & 255;
    size_t tb = ((size_t)(c * H + h) * D + d) * BLK + j;
    kTout[tb] = f2bf(ko1); kTout[tb + 64 * BLK] = f2bf(ko2);
    vTout[tb] = v1;        vTout[tb + 64 * BLK] = v2;
}

// ---------------------------------------------------------------------------
// Phase A: KV[c][h] = (k ⊙ k_decay)^T @ v  (128x128 f32, K=256)
// ---------------------------------------------------------------------------
__global__ __launch_bounds__(256) void phaseA(const u16* __restrict__ kT,
                                              const u16* __restrict__ vT,
                                              float* __restrict__ KV) {
    int b = blockIdx.x;
    int c = b >> 4, h = b & 15;
    float s = slope_h(h);
    __shared__ u16 ldsA[128 * 40];
    __shared__ u16 ldsB[128 * 40];
    const int tid = threadIdx.x;
    const int wave = tid >> 5, lane = tid & 31, l16 = lane & 15;
    const int khalf = (lane >> 4) & 1;
    const int wm = (wave >> 2) * 64, wn = (wave & 3) * 32;

    f32x8 acc[4][2] = {};
    size_t chbase = (size_t)(c * H + h) * (D * BLK);
    const int lrow = tid >> 1, lcol = (tid & 1) * 16;

    for (int k0 = 0; k0 < BLK; k0 += 32) {
        alignas(16) u16 abuf[16];
        const u16* gk = kT + chbase + (size_t)lrow * BLK + k0 + lcol;
#pragma unroll
        for (int e = 0; e < 16; ++e) {
            int j = k0 + lcol + e;
            abuf[e] = f2bf(bf2f(gk[e]) * __expf(-s * (float)(BLK - 1 - j)));
        }
        const uint4* gv = (const uint4*)(vT + chbase + (size_t)lrow * BLK + k0 + lcol);
        uint4 v0 = gv[0], v1 = gv[1];
        __syncthreads();
        uint4* sa4 = (uint4*)(ldsA + lrow * 40 + lcol);
        sa4[0] = ((const uint4*)abuf)[0];
        sa4[1] = ((const uint4*)abuf)[1];
        uint4* sb4 = (uint4*)(ldsB + lrow * 40 + lcol);
        sb4[0] = v0; sb4[1] = v1;
        __syncthreads();

        bf16x16 bfr[2];
#pragma unroll
        for (int j = 0; j < 2; ++j)
            bfr[j] = load_fragB(ldsB + (wn + j * 16 + l16) * 40 + (khalf ? 16 : 0));
#pragma unroll
        for (int i = 0; i < 4; ++i) {
            bf16x16 afr = load_fragA(ldsA + (wm + i * 16 + l16) * 40 + (khalf ? 16 : 0));
#pragma unroll
            for (int j = 0; j < 2; ++j)
                acc[i][j] = wmma_bf16(afr, bfr[j], acc[i][j]);
        }
    }
    size_t ob = (size_t)(c * H + h) * (D * D);
#pragma unroll
    for (int i = 0; i < 4; ++i)
#pragma unroll
        for (int j = 0; j < 2; ++j)
#pragma unroll
            for (int r = 0; r < 8; ++r) {
                int row = wm + i * 16 + r + khalf * 8;
                int col = wn + j * 16 + l16;
                KV[ob + row * D + col] = acc[i][j][r];
            }
}

// ---------------------------------------------------------------------------
// Phase B: per-head prefix scan of 128x128 states over 16 chunks (emit S^T bf16)
// ---------------------------------------------------------------------------
__global__ __launch_bounds__(256) void phaseB(const float* __restrict__ KV,
                                              u16* __restrict__ StT) {
    int h = blockIdx.x, tid = threadIdx.x;
    float s = slope_h(h);
    float bd = __expf(-s * (float)BLK);
    float S[64];
#pragma unroll
    for (int j = 0; j < 64; ++j) S[j] = 0.f;
    for (int c = 0; c < NC; ++c) {
        size_t base = (size_t)(c * H + h) * (D * D);
#pragma unroll
        for (int j = 0; j < 64; ++j) {
            int i = tid + j * 256;
            int d = i >> 7, e = i & 127;
            StT[base + e * D + d] = f2bf(S[j]);
            S[j] = bd * S[j] + KV[base + i];
        }
    }
}

// ---------------------------------------------------------------------------
// Phase C: o = mask(q k^T) v + q_decay ⊙ (q @ S_prefix), per (chunk, head).
// Tile staging via TDM (tensor_load_to_lds) or async-LDS copies.
// ---------------------------------------------------------------------------
__global__ __launch_bounds__(256) void phaseC(const u16* __restrict__ q,
                                              const u16* __restrict__ k,
                                              const u16* __restrict__ vT,
                                              const u16* __restrict__ StT,
                                              float* __restrict__ o) {
    int b = blockIdx.x;
    int c = b >> 4, h = b & 15;
    float s = slope_h(h);
    __shared__ u16 ldsSt[128 * 136];
    __shared__ u16 ldsK[64 * 136];
    __shared__ u16 ldsVT[128 * 72];
    __shared__ u16 ldsP[8][32 * 40];

    const int tid = threadIdx.x;
    const int wave = tid >> 5, lane = tid & 31, l16 = lane & 15;
    const int khalf = (lane >> 4) & 1;
    const int mbase = wave * 32;

    const u16* stg = StT + (size_t)(c * H + h) * (D * D);
#if TRY_TDM
    if (wave == 0) {
        // 128x128 bf16 tile, row stride 128, LDS pad 4 dwords per 64 dwords -> 136
        tdm_load_2d_bf16(lds_off_u32(ldsSt), stg, 128, 128, 128, 5, 3);
        tensor_wait0();
    }
#else
    {
        const u16* g = stg + (tid >> 1) * D + (tid & 1) * 64;
        u16* sp = ldsSt + (tid >> 1) * 136 + (tid & 1) * 64;
#pragma unroll
        for (int q4 = 0; q4 < 8; ++q4)
            async_copy16(g + q4 * 8, sp + q4 * 8);
        async_join();
    }
#endif
    // q fragments (kept in registers, reused for inter + intra)
    bf16x16 qf[2][4];
    const u16* qrowbase = q + (size_t)(c * BLK) * HID + h * D;
#pragma unroll
    for (int mi = 0; mi < 2; ++mi) {
        int row = mbase + mi * 16 + l16;
        const u16* p = qrowbase + (size_t)row * HID;
#pragma unroll
        for (int kc = 0; kc < 4; ++kc)
            qf[mi][kc] = load_fragA(p + kc * 32 + (khalf ? 16 : 0));
    }
    __syncthreads();

    f32x8 acc[2][8] = {};
    // inter: q @ S  (Bt = S^T in LDS)
#pragma unroll
    for (int nt = 0; nt < 8; ++nt)
#pragma unroll
        for (int kc = 0; kc < 4; ++kc) {
            bf16x16 bf = load_fragB(ldsSt + (nt * 16 + l16) * 136 + kc * 32 + (khalf ? 16 : 0));
#pragma unroll
            for (int mi = 0; mi < 2; ++mi)
                acc[mi][nt] = wmma_bf16(qf[mi][kc], bf, acc[mi][nt]);
        }
    // row-scale by q_decay(i) = exp(-s*(i+1))
#pragma unroll
    for (int mi = 0; mi < 2; ++mi)
#pragma unroll
        for (int r = 0; r < 8; ++r) {
            float qd = __expf(-s * (float)(mbase + mi * 16 + r + khalf * 8 + 1));
#pragma unroll
            for (int nt = 0; nt < 8; ++nt) acc[mi][nt][r] *= qd;
        }

    // intra: stages of 64 columns j
    const u16* krowbase = k + (size_t)(c * BLK) * HID + h * D;
    const u16* vTbase = vT + (size_t)(c * H + h) * (D * BLK);
    for (int js = 0; js < 4; ++js) {
        int j0 = js * 64;
        __syncthreads();
#if TRY_TDM
        if (wave == 0) {
            // k rows j0..j0+63 (128 wide, stride HID) -> ldsK (pad to 136)
            tdm_load_2d_bf16(lds_off_u32(ldsK), krowbase + (size_t)j0 * HID,
                             128, 64, HID, 5, 3);
            // v^T rows d=0..127, cols j0..j0+63 (stride BLK) -> ldsVT (pad to 72)
            tdm_load_2d_bf16(lds_off_u32(ldsVT), vTbase + j0,
                             64, 128, BLK, 4, 3);
            tensor_wait0();
        }
#else
        {
            int jj = tid >> 2, seg = (tid & 3) * 32;
            const u16* gp = krowbase + (size_t)(j0 + jj) * HID + seg;
            u16* sp = ldsK + jj * 136 + seg;
#pragma unroll
            for (int e = 0; e < 4; ++e) async_copy16(gp + e * 8, sp + e * 8);
        }
        {
            int d = tid >> 1, seg = (tid & 1) * 32;
            const u16* gp = vTbase + (size_t)d * BLK + j0 + seg;
            u16* sp = ldsVT + d * 72 + seg;
#pragma unroll
            for (int e = 0; e < 4; ++e) async_copy16(gp + e * 8, sp + e * 8);
        }
        async_join();
#endif
        __syncthreads();

        for (int jp = 0; jp < 2; ++jp) {
            int pglob = js * 2 + jp;            // 32-col pair index 0..7
            if (pglob > wave) continue;         // wave-uniform causal skip

            // P = q k^T for 2 m-tiles x 2 j-tiles (K = 128 over d)
            f32x8 pacc[2][2] = {};
#pragma unroll
            for (int ji = 0; ji < 2; ++ji) {
                int jrow = jp * 32 + ji * 16 + l16;
#pragma unroll
                for (int kc = 0; kc < 4; ++kc) {
                    bf16x16 kf = load_fragB(ldsK + jrow * 136 + kc * 32 + (khalf ? 16 : 0));
#pragma unroll
                    for (int mi = 0; mi < 2; ++mi)
                        pacc[mi][ji] = wmma_bf16(qf[mi][kc], kf, pacc[mi][ji]);
                }
            }
            // mask + decay in f32, re-fragment P through per-wave LDS
            u16* pbuf = ldsP[wave];
#pragma unroll
            for (int mi = 0; mi < 2; ++mi)
#pragma unroll
                for (int ji = 0; ji < 2; ++ji) {
                    int jcol = j0 + jp * 32 + ji * 16 + l16;
#pragma unroll
                    for (int r = 0; r < 8; ++r) {
                        int i = mbase + mi * 16 + r + khalf * 8;
                        int diff = i - jcol;
                        float v = (diff >= 0) ? pacc[mi][ji][r] * __expf(-s * (float)diff)
                                              : 0.f;
                        pbuf[(mi * 16 + r + khalf * 8) * 40 + ji * 16 + l16] = f2bf(v);
                    }
                }
            // acc += P @ v  (Bt = v^T in LDS, K = 32 over j)
#pragma unroll
            for (int mi = 0; mi < 2; ++mi) {
                bf16x16 pf = load_fragA(pbuf + (mi * 16 + l16) * 40 + (khalf ? 16 : 0));
#pragma unroll
                for (int nt = 0; nt < 8; ++nt) {
                    bf16x16 vf = load_fragB(ldsVT + (nt * 16 + l16) * 72 + jp * 32 + (khalf ? 16 : 0));
                    acc[mi][nt] = wmma_bf16(pf, vf, acc[mi][nt]);
                }
            }
        }
    }
#pragma unroll
    for (int mi = 0; mi < 2; ++mi)
#pragma unroll
        for (int nt = 0; nt < 8; ++nt) {
            int col = h * D + nt * 16 + l16;
#pragma unroll
            for (int r = 0; r < 8; ++r) {
                int row = c * BLK + mbase + mi * 16 + r + khalf * 8;
                o[(size_t)row * HID + col] = acc[mi][nt][r];
            }
        }
}

// ---------------------------------------------------------------------------
// RMS-norm over H*D + sigmoid gate, emit bf16 for final GEMM
// ---------------------------------------------------------------------------
__global__ __launch_bounds__(256) void normgate(const float* __restrict__ o,
                                                const u16* __restrict__ gatepre,
                                                const float* __restrict__ gw,
                                                u16* __restrict__ yin) {
    int t = blockIdx.x, tid = threadIdx.x;
    __shared__ float red[256];
    const float* orow = o + (size_t)t * HID;
    float vals[8];
    float ss = 0.f;
#pragma unroll
    for (int j = 0; j < 8; ++j) {
        float v = orow[tid + j * 256];
        vals[j] = v;
        ss += v * v;
    }
    red[tid] = ss;
    __syncthreads();
    for (int st = 128; st > 0; st >>= 1) {
        if (tid < st) red[tid] += red[tid + st];
        __syncthreads();
    }
    float inv = rsqrtf(red[0] / (float)HID + 1e-5f);
    const u16* grow = gatepre + (size_t)t * HID;
    u16* yrow = yin + (size_t)t * HID;
#pragma unroll
    for (int j = 0; j < 8; ++j) {
        int col = tid + j * 256;
        float on = vals[j] * inv * gw[col];
        float g = 1.f / (1.f + __expf(-bf2f(grow[col])));
        yrow[col] = f2bf(on * g);
    }
}

// ---------------------------------------------------------------------------
// Launch
// ---------------------------------------------------------------------------
extern "C" void kernel_launch(void* const* d_in, const int* in_sizes, int n_in,
                              void* d_out, int out_size, void* d_ws, size_t ws_size,
                              hipStream_t stream) {
    const float* hs   = (const float*)d_in[0];
    const int*   pos  = (const int*)d_in[1];
    const float* wqkv = (const float*)d_in[2];
    const float* wg   = (const float*)d_in[3];
    const float* wd   = (const float*)d_in[4];
    const float* gnw  = (const float*)d_in[5];

    char* ws = (char*)d_ws;
    const size_t MB = 1u << 20;
    u16*   hsb   = (u16*)(ws + 0 * MB);     // 16 MiB  [T][HID] bf16
    u16*   wqkvT = (u16*)(ws + 16 * MB);    // 24 MiB  [6144][2048]
    u16*   wgT   = (u16*)(ws + 40 * MB);    //  8 MiB  [2048][2048]
    u16*   wdT   = (u16*)(ws + 48 * MB);    //  8 MiB  [2048][2048]
    u16*   qkvb  = (u16*)(ws + 56 * MB);    // 48 MiB  [T][6144] (dead after rope)
    u16*   gateb = (u16*)(ws + 104 * MB);   // 16 MiB  [T][HID]
    u16*   qb    = (u16*)(ws + 120 * MB);   // 16 MiB
    u16*   kb    = (u16*)(ws + 136 * MB);   // 16 MiB
    u16*   kTb   = (u16*)(ws + 152 * MB);   // 16 MiB  [NC][H][D][BLK]
    u16*   vTb   = (u16*)(ws + 168 * MB);   // 16 MiB  [NC][H][D][BLK]
    float* KV    = (float*)(ws + 184 * MB); // 16 MiB  [NC][H][D][D] f32
    u16*   StT   = (u16*)(ws + 200 * MB);   //  8 MiB  [NC][H][D][D] bf16
    float* obuf  = (float*)(ws + 56 * MB);  // 32 MiB  alias of qkvb region
    u16*   yin   = (u16*)(ws + 88 * MB);    // 16 MiB  alias of qkvb region tail

    cvt_bf16<<<(T * HID / 4 + 255) / 256, 256, 0, stream>>>(hs, hsb, T * HID);
    transpose_cvt<<<dim3((3 * H * D) / 32, HID / 32), 256, 0, stream>>>(wqkv, wqkvT, HID, 3 * H * D);
    transpose_cvt<<<dim3(HID / 32, HID / 32), 256, 0, stream>>>(wg, wgT, HID, HID);
    transpose_cvt<<<dim3(HID / 32, HID / 32), 256, 0, stream>>>(wd, wdT, HID, HID);

    gemm_nt<true><<<dim3((3 * H * D) / 128, T / 128), 256, 0, stream>>>(
        hsb, wqkvT, (void*)qkvb, T, 3 * H * D, HID);
    gemm_nt<true><<<dim3(HID / 128, T / 128), 256, 0, stream>>>(
        hsb, wgT, (void*)gateb, T, HID, HID);

    rope_pack<<<(T * H * 64) / 256, 256, 0, stream>>>(qkvb, pos, qb, kb, kTb, vTb);

    phaseA<<<NC * H, 256, 0, stream>>>(kTb, vTb, KV);
    phaseB<<<H, 256, 0, stream>>>(KV, StT);
    phaseC<<<NC * H, 256, 0, stream>>>(qb, kb, vTb, StT, obuf);

    normgate<<<T, 256, 0, stream>>>(obuf, gateb, gnw, yin);
    gemm_nt<false><<<dim3(HID / 128, T / 128), 256, 0, stream>>>(
        yin, wdT, d_out, T, HID, HID);
}
